// ESIM_8796093022741
// MI455X (gfx1250) — matrix operand order repaired and checked by reading, not compiled, not run on hardware
//
#include <hip/hip_runtime.h>
#include <cstddef>
#include <cstdint>

typedef unsigned short u16;
typedef __attribute__((ext_vector_type(8)))  u16    v8u16;
typedef __attribute__((ext_vector_type(16))) __bf16 v16bf;
typedef __attribute__((ext_vector_type(8)))  float  v8f;

static constexpr int   B_   = 128;
static constexpr int   T_   = 64;
static constexpr int   E_   = 300;
static constexpr int   Ep_  = 320;
static constexpr int   H_   = 300;
static constexpr int   Hp_  = 320;
static constexpr int   G_   = 1200;   // 4*H (gate width)
static constexpr int   Gp_  = 1216;   // gate width padded to x64 for LDS-tiled GEMM
static constexpr int   D2_  = 600;    // 2*H
static constexpr int   D2p_ = 608;    // padded to x32 for WMMA K
static constexpr int   M8_  = 2400;   // 8*H
static constexpr int   UN_  = 512;
static constexpr float NEGF = -4294967295.0f;

__device__ __forceinline__ u16 f2bf(float f) {
  unsigned x = __float_as_uint(f);
  unsigned r = x + 0x7FFFu + ((x >> 16) & 1u);   // round-to-nearest-even
  return (u16)(r >> 16);
}
__device__ __forceinline__ float sigm(float x) { return 1.0f / (1.0f + __expf(-x)); }

union FragAB { v8u16 h[2]; v16bf v; };

// Per-lane async copy of 16 bytes global -> LDS (ASYNCcnt-tracked).
__device__ __forceinline__ void async_b128(unsigned lds_off, const u16* gptr) {
  asm volatile("global_load_async_to_lds_b128 %0, %1, off"
               :: "v"(lds_off), "v"((unsigned long long)(uintptr_t)gptr)
               : "memory");
}
__device__ __forceinline__ void wait_async0() {
  asm volatile("s_wait_asynccnt 0x0" ::: "memory");
}

// ---------------------------------------------------------------------------
// LDS-staged bf16 WMMA GEMM. C = A(MxKp) @ Bt(NpadxKp)^T.
// Macro tile 64x64, 4 waves/block, wave w -> rows [w*16,w*16+16), 4 N-tiles.
// A 64x32 and B 64x32 K-slabs staged in LDS via GLOBAL_LOAD_ASYNC_TO_LDS_B128;
// 80-byte LDS row stride (16B-aligned, bank-conflict-free fragment reads).
// mode 0: none | 1: +bias[N] | 2: +Cin[MxN](ldci) | 3: +bias, tanh
// ---------------------------------------------------------------------------
#define LROW 40   // u16 per LDS row (80 bytes)

__global__ __launch_bounds__(128)
void gemm_bf16_wmma_lds(const u16* __restrict__ A, int lda,
                        const u16* __restrict__ Bt, int Kp,
                        const float* __restrict__ Cin, int ldci,
                        float* __restrict__ C, int ldc,
                        int M, int Nreal, int nb64, int mode)
{
  __shared__ u16 ldsA[64 * LROW];
  __shared__ u16 ldsB[64 * LROW];

  const int tid  = threadIdx.x;
  const int lane = tid & 31;
  const int wave = tid >> 5;
  const int bm   = blockIdx.x / nb64;
  const int bn   = blockIdx.x % nb64;

  const unsigned ldsAb = (unsigned)(uintptr_t)(void*)ldsA;
  const unsigned ldsBb = (unsigned)(uintptr_t)(void*)ldsB;

  // Staging plan: 256 16B chunks per matrix; this thread owns chunks 2t, 2t+1.
  const int c0  = tid * 2;
  const int ra0 = c0 >> 2,       qa0 = c0 & 3;
  const int ra1 = (c0 + 1) >> 2, qa1 = (c0 + 1) & 3;

  const u16* Arow0 = A  + (long long)(bm * 64 + ra0) * lda + qa0 * 8;
  const u16* Arow1 = A  + (long long)(bm * 64 + ra1) * lda + qa1 * 8;
  const u16* Brow0 = Bt + (long long)(bn * 64 + ra0) * Kp  + qa0 * 8;
  const u16* Brow1 = Bt + (long long)(bn * 64 + ra1) * Kp  + qa1 * 8;
  const unsigned la0 = ldsAb + (unsigned)(ra0 * 80 + qa0 * 16);
  const unsigned la1 = ldsAb + (unsigned)(ra1 * 80 + qa1 * 16);
  const unsigned lb0 = ldsBb + (unsigned)(ra0 * 80 + qa0 * 16);
  const unsigned lb1 = ldsBb + (unsigned)(ra1 * 80 + qa1 * 16);

  v8f acc[4];
#pragma unroll
  for (int nt = 0; nt < 4; ++nt) acc[nt] = (v8f){0.f,0.f,0.f,0.f,0.f,0.f,0.f,0.f};

  const int arow = wave * 16 + (lane & 15);
  const unsigned aoff = (unsigned)(arow * LROW + ((lane >> 4) << 4));  // u16 units

  for (int k0 = 0; k0 < Kp; k0 += 32) {
    async_b128(la0, Arow0 + k0);
    async_b128(la1, Arow1 + k0);
    async_b128(lb0, Brow0 + k0);
    async_b128(lb1, Brow1 + k0);
    wait_async0();
    __syncthreads();

    FragAB fa;
    fa.h[0] = *(const v8u16*)&ldsA[aoff];
    fa.h[1] = *(const v8u16*)&ldsA[aoff + 8];
#pragma unroll
    for (int nt = 0; nt < 4; ++nt) {
      FragAB fb;
      const unsigned boff = (unsigned)((nt * 16 + (lane & 15)) * LROW + ((lane >> 4) << 4));
      fb.h[0] = *(const v8u16*)&ldsB[boff];
      fb.h[1] = *(const v8u16*)&ldsB[boff + 8];
      acc[nt] = __builtin_amdgcn_wmma_f32_16x16x32_bf16(false, fa.v, false, fb.v,
                                                        (short)0, acc[nt], false, false);
    }
    __syncthreads();   // protect LDS before next stage overwrites
  }

  const int mrow = bm * 64 + wave * 16 + ((lane >> 4) << 3);
#pragma unroll
  for (int nt = 0; nt < 4; ++nt) {
    const int nbase = bn * 64 + nt * 16;
    if (nbase < Nreal) {                      // wave-uniform edge guard
      const int ncol = nbase + (lane & 15);
      const float bv = (mode == 1 || mode == 3) ? Cin[ncol] : 0.f;
#pragma unroll
      for (int v = 0; v < 8; ++v) {
        float x = acc[nt][v] + bv;
        if (mode == 2) x += Cin[(long long)(mrow + v) * ldci + ncol];
        if (mode == 3) x = tanhf(x);
        C[(long long)(mrow + v) * ldc + ncol] = x;
      }
    }
  }
}

// ---------------------------------------------------------------------------
// Direct-global batched WMMA GEMM (kept for the small batched attention GEMMs)
// ---------------------------------------------------------------------------
__global__ __launch_bounds__(128)
void gemm_bf16_wmma(const u16* __restrict__ A, long long strideA, int lda,
                    const u16* __restrict__ Bt, long long strideB,
                    const float* __restrict__ Cin, long long strideCin, int ldci,
                    float* __restrict__ C, long long strideC, int ldc,
                    int M, int N, int Kp, int mode)
{
  const int lane   = threadIdx.x;
  const int tilesN = N >> 4;
  const int tile   = blockIdx.x * 4 + threadIdx.y;
  if (tile >= (M >> 4) * tilesN) return;          // wave-uniform
  const int tm = tile / tilesN;
  const int tn = tile % tilesN;

  const u16* Ab = A  + (long long)blockIdx.y * strideA;
  const u16* Bb = Bt + (long long)blockIdx.y * strideB;

  const u16* ap = Ab + (long long)((tm << 4) + (lane & 15)) * lda + ((lane >> 4) << 4);
  const u16* bp = Bb + (long long)((tn << 4) + (lane & 15)) * Kp + ((lane >> 4) << 4);

  v8f acc = {0.f, 0.f, 0.f, 0.f, 0.f, 0.f, 0.f, 0.f};
  for (int k0 = 0; k0 < Kp; k0 += 32) {
    FragAB fa, fb;
    fa.h[0] = *(const v8u16*)(ap + k0);
    fa.h[1] = *(const v8u16*)(ap + k0 + 8);
    fb.h[0] = *(const v8u16*)(bp + k0);
    fb.h[1] = *(const v8u16*)(bp + k0 + 8);
    __builtin_prefetch(ap + k0 + 32, 0, 0);
    __builtin_prefetch(bp + k0 + 32, 0, 0);
    acc = __builtin_amdgcn_wmma_f32_16x16x32_bf16(false, fa.v, false, fb.v,
                                                  (short)0, acc, false, false);
  }

  const int mrow = (tm << 4) + ((lane >> 4) << 3);
  const int ncol = (tn << 4) + (lane & 15);
  float*       Cb = C + (long long)blockIdx.y * strideC;
  const float* Ci = Cin ? (Cin + (long long)blockIdx.y * strideCin) : nullptr;
#pragma unroll
  for (int v = 0; v < 8; ++v) {
    float x = acc[v];
    if (mode == 1 || mode == 3) x += Ci[ncol];
    else if (mode == 2)         x += Ci[(long long)(mrow + v) * ldci + ncol];
    if (mode == 3) x = tanhf(x);
    Cb[(long long)(mrow + v) * ldc + ncol] = x;
  }
}

// ---------------------------------------------------------------------------
// Helper kernels
// ---------------------------------------------------------------------------
__global__ void zero16_k(u16* p, long long n) {
  long long i = (long long)blockIdx.x * blockDim.x + threadIdx.x;
  if (i < n) p[i] = 0;
}
__global__ void zero32_k(float* p, long long n) {
  long long i = (long long)blockIdx.x * blockDim.x + threadIdx.x;
  if (i < n) p[i] = 0.f;
}

__global__ void gather_emb_k(const int* __restrict__ idx, const float* __restrict__ Wemb,
                             u16* __restrict__ dst) {
  long long i = (long long)blockIdx.x * blockDim.x + threadIdx.x;
  if (i >= (long long)B_ * T_ * Ep_) return;
  int r = (int)(i / Ep_), c = (int)(i % Ep_);
  int w = idx[r];
  dst[i] = (c < E_) ? f2bf(Wemb[(long long)w * E_ + c]) : (u16)0;
}

// dst[b][n][k] = (k<srows && n<scols) ? bf16(src[b][k][n]) : 0
__global__ void transpose_cvt_k(const float* __restrict__ src, long long sstride,
                                int srows, int scols,
                                u16* __restrict__ dst, long long dstride,
                                int Np, int Kp) {
  long long i = (long long)blockIdx.x * blockDim.x + threadIdx.x;
  if (i >= (long long)Np * Kp) return;
  int n = (int)(i / Kp), k = (int)(i % Kp);
  const float* sb = src + (long long)blockIdx.y * sstride;
  u16*         db = dst + (long long)blockIdx.y * dstride;
  db[i] = (k < srows && n < scols) ? f2bf(sb[(long long)k * scols + n]) : (u16)0;
}

__global__ void cvt_pad_k(const float* __restrict__ src, int lds,
                          u16* __restrict__ dst, int C, int Cp, long long total) {
  long long i = (long long)blockIdx.x * blockDim.x + threadIdx.x;
  if (i >= total) return;
  int r = (int)(i / Cp), c = (int)(i % Cp);
  dst[i] = (c < C) ? f2bf(src[(long long)r * lds + c]) : (u16)0;
}

__global__ void lstm_cell_k(const float* __restrict__ z, float* __restrict__ c,
                            float* __restrict__ hseq, int col_off, int t,
                            u16* __restrict__ hnext, int actRelu) {
  int i = blockIdx.x * blockDim.x + threadIdx.x;
  if (i >= B_ * H_) return;
  int b = i / H_, j = i % H_;
  const float* zr = z + (long long)b * G_;
  float zi = zr[j], zf = zr[H_ + j], zg = zr[2 * H_ + j], zo = zr[3 * H_ + j];
  float g  = actRelu ? fmaxf(zg, 0.f) : tanhf(zg);
  float cn = sigm(zf) * c[i] + sigm(zi) * g;
  c[i] = cn;
  float ca = actRelu ? fmaxf(cn, 0.f) : tanhf(cn);
  float h  = sigm(zo) * ca;
  hseq[((long long)b * T_ + t) * D2_ + col_off + j] = h;
  hnext[b * Hp_ + j] = f2bf(h);
}

__global__ void softmax_aw_k(const float* __restrict__ e, const int* __restrict__ h_index,
                             float* __restrict__ aw) {
  __shared__ float buf[T_];
  int row = blockIdx.x;            // b*T + p
  int b = row >> 6;
  int q = threadIdx.x;
  float x = e[(long long)row * T_ + q] + (h_index[b * T_ + q] == 0 ? NEGF : 0.f);
  buf[q] = x; __syncthreads();
  for (int s = 32; s > 0; s >>= 1) { if (q < s) buf[q] = fmaxf(buf[q], buf[q + s]); __syncthreads(); }
  float m = buf[0]; __syncthreads();
  float ex = __expf(x - m);
  buf[q] = ex; __syncthreads();
  for (int s = 32; s > 0; s >>= 1) { if (q < s) buf[q] += buf[q + s]; __syncthreads(); }
  aw[(long long)row * T_ + q] = ex / buf[0];
}

__global__ void softmax_bw_k(const float* __restrict__ e, const int* __restrict__ p_index,
                             float* __restrict__ bw) {
  __shared__ float buf[T_];
  int row = blockIdx.x;            // b*T + q
  int b = row >> 6, q = row & 63;
  int p = threadIdx.x;
  float x = e[((long long)b * T_ + p) * T_ + q] + (p_index[b * T_ + p] == 0 ? NEGF : 0.f);
  buf[p] = x; __syncthreads();
  for (int s = 32; s > 0; s >>= 1) { if (p < s) buf[p] = fmaxf(buf[p], buf[p + s]); __syncthreads(); }
  float m = buf[0]; __syncthreads();
  float ex = __expf(x - m);
  buf[p] = ex; __syncthreads();
  for (int s = 32; s > 0; s >>= 1) { if (p < s) buf[p] += buf[p + s]; __syncthreads(); }
  bw[(long long)row * T_ + p] = ex / buf[0];
}

__global__ void build_m_k(const float* __restrict__ aMat /*[r,608]*/,
                          const float* __restrict__ xseq /*[r,600]*/,
                          u16* __restrict__ m /*[r,2400]*/) {
  long long i = (long long)blockIdx.x * blockDim.x + threadIdx.x;
  if (i >= (long long)B_ * T_ * D2_) return;
  int r = (int)(i / D2_), j = (int)(i % D2_);
  float a = aMat[(long long)r * D2p_ + j];
  float x = xseq[(long long)r * D2_ + j];
  u16* mr = m + (long long)r * M8_;
  mr[j]            = f2bf(a);
  mr[D2_ + j]      = f2bf(x);
  mr[2 * D2_ + j]  = f2bf(a - x);
  mr[3 * D2_ + j]  = f2bf(a * x);
}

__global__ void pool_meanmax_k(const float* __restrict__ seq /*[B,T,600]*/,
                               float* __restrict__ v, int colbase) {
  int i = blockIdx.x * blockDim.x + threadIdx.x;
  if (i >= B_ * D2_) return;
  int b = i / D2_, d = i % D2_;
  const float* s = seq + ((long long)b * T_) * D2_ + d;
  float mx = -3.4e38f, sm = 0.f;
  for (int t = 0; t < T_; ++t) { float x = s[(long long)t * D2_]; sm += x; mx = fmaxf(mx, x); }
  v[(long long)b * M8_ + colbase + d]       = sm * (1.f / T_);
  v[(long long)b * M8_ + colbase + D2_ + d] = mx;
}

__global__ void fc2_k(const float* __restrict__ x /*[128,512]*/,
                      const float* __restrict__ W2 /*[512,2]*/,
                      const float* __restrict__ b2, float* __restrict__ out) {
  int i = blockIdx.x * blockDim.x + threadIdx.x;
  if (i >= B_ * 2) return;
  int b = i >> 1, o = i & 1;
  float s = b2[o];
  for (int k = 0; k < UN_; ++k) s += x[b * UN_ + k] * W2[k * 2 + o];
  out[b * 2 + o] = tanhf(s);
}

// ---------------------------------------------------------------------------
// Host orchestration
// ---------------------------------------------------------------------------
struct WsAlloc {
  char* base; size_t off;
  void* get(size_t bytes) {
    off = (off + 255) & ~(size_t)255;
    void* p = base + off; off += bytes; return p;
  }
};

static inline void launch_gemm_lds(hipStream_t s, const u16* A, int lda,
                                   const u16* Bt, int Kp, const float* Cin, int ldci,
                                   float* C, int ldc, int M, int N, int mode) {
  int nb64 = (N + 63) / 64;
  gemm_bf16_wmma_lds<<<(M / 64) * nb64, 128, 0, s>>>(A, lda, Bt, Kp, Cin, ldci,
                                                     C, ldc, M, N, nb64, mode);
}

static inline void launch_gemm(hipStream_t s,
                               const u16* A, long long sA, int lda,
                               const u16* Bt, long long sB,
                               const float* Cin, long long sCi, int ldci,
                               float* C, long long sC, int ldc,
                               int M, int N, int Kp, int mode, int batch) {
  int tiles = (M / 16) * (N / 16);
  dim3 grid((tiles + 3) / 4, batch), block(32, 4);
  gemm_bf16_wmma<<<grid, block, 0, s>>>(A, sA, lda, Bt, sB, Cin, sCi, ldci,
                                        C, sC, ldc, M, N, Kp, mode);
}

static inline void zero16(hipStream_t s, u16* p, long long n) {
  zero16_k<<<(unsigned)((n + 255) / 256), 256, 0, s>>>(p, n);
}
static inline void zero32(hipStream_t s, float* p, long long n) {
  zero32_k<<<(unsigned)((n + 255) / 256), 256, 0, s>>>(p, n);
}

static void run_lstm(hipStream_t s, const float* xz /*[B,T,1200]*/,
                     const u16* WhT /*[1216,320] bf16*/,
                     float* hseq, int col_off, int reverse, int relu,
                     float* z, u16* hping, u16* hpong, float* c) {
  zero16(s, hping, (long long)B_ * Hp_);
  zero16(s, hpong, (long long)B_ * Hp_);
  zero32(s, c, (long long)B_ * H_);
  u16* hprev = hping; u16* hnext = hpong;
  for (int step = 0; step < T_; ++step) {
    int t = reverse ? (T_ - 1 - step) : step;
    // z = hprev @ Wh + xz[:, t, :]  (LDS-staged WMMA, mode 2)
    launch_gemm_lds(s, hprev, Hp_, WhT, Hp_, xz + (long long)t * G_, T_ * G_,
                    z, G_, B_, G_, /*mode=*/2);
    lstm_cell_k<<<(B_ * H_ + 255) / 256, 256, 0, s>>>(z, c, hseq, col_off, t, hnext, relu);
    u16* tmp = hprev; hprev = hnext; hnext = tmp;
  }
}

extern "C" void kernel_launch(void* const* d_in, const int* in_sizes, int n_in,
                              void* d_out, int out_size, void* d_ws, size_t ws_size,
                              hipStream_t stream) {
  (void)in_sizes; (void)n_in; (void)out_size; (void)ws_size;
  const int*   p_index = (const int*)d_in[0];
  const int*   h_index = (const int*)d_in[1];
  const float* W_emb   = (const float*)d_in[2];
  const float *Wx1[4], *Wh1[4], *bb1[4], *Wx2[4], *Wh2[4], *bb2[4];
  for (int i = 0; i < 4; ++i) {
    Wx1[i] = (const float*)d_in[3 + 3 * i];
    Wh1[i] = (const float*)d_in[4 + 3 * i];
    bb1[i] = (const float*)d_in[5 + 3 * i];
    Wx2[i] = (const float*)d_in[15 + 3 * i];
    Wh2[i] = (const float*)d_in[16 + 3 * i];
    bb2[i] = (const float*)d_in[17 + 3 * i];
  }
  const float* W1  = (const float*)d_in[27];
  const float* b1  = (const float*)d_in[28];
  const float* W2  = (const float*)d_in[29];
  const float* b2v = (const float*)d_in[30];
  float* out = (float*)d_out;

  WsAlloc ws{(char*)d_ws, 0};
  const long long R = (long long)B_ * T_;   // 8192 rows

  u16*   embP  = (u16*)ws.get(R * Ep_ * 2);
  u16*   embH  = (u16*)ws.get(R * Ep_ * 2);
  u16*   WxT1[4], *WhT1[4], *WxT2[4], *WhT2[4];
  for (int i = 0; i < 4; ++i) WxT1[i] = (u16*)ws.get((long long)Gp_ * Hp_ * 2);
  for (int i = 0; i < 4; ++i) WhT1[i] = (u16*)ws.get((long long)Gp_ * Hp_ * 2);
  for (int i = 0; i < 4; ++i) WxT2[i] = (u16*)ws.get((long long)Gp_ * M8_ * 2);
  for (int i = 0; i < 4; ++i) WhT2[i] = (u16*)ws.get((long long)Gp_ * Hp_ * 2);
  u16*   W1T   = (u16*)ws.get((long long)UN_ * M8_ * 2);
  float* xz    = (float*)ws.get(R * G_ * 4);
  float* z     = (float*)ws.get((long long)B_ * G_ * 4);
  u16*   hping = (u16*)ws.get((long long)B_ * Hp_ * 2);
  u16*   hpong = (u16*)ws.get((long long)B_ * Hp_ * 2);
  float* cbuf  = (float*)ws.get((long long)B_ * H_ * 4);
  float* pseq  = (float*)ws.get(R * D2_ * 4);
  float* hseq  = (float*)ws.get(R * D2_ * 4);
  float* a2seq = (float*)ws.get(R * D2_ * 4);
  float* b2seq = (float*)ws.get(R * D2_ * 4);
  u16*   p_bf  = (u16*)ws.get(R * D2p_ * 2);
  u16*   h_bf  = (u16*)ws.get(R * D2p_ * 2);
  float* emat  = (float*)ws.get((long long)B_ * T_ * T_ * 4);
  float* aw    = (float*)ws.get((long long)B_ * T_ * T_ * 4);
  float* bw    = (float*)ws.get((long long)B_ * T_ * T_ * 4);
  u16*   aw_bf = (u16*)ws.get((long long)B_ * T_ * T_ * 2);
  u16*   bw_bf = (u16*)ws.get((long long)B_ * T_ * T_ * 2);
  u16*   hT_bf = (u16*)ws.get((long long)B_ * D2p_ * T_ * 2);
  u16*   pT_bf = (u16*)ws.get((long long)B_ * D2p_ * T_ * 2);
  float* aMat  = (float*)ws.get(R * D2p_ * 4);
  float* bMat  = (float*)ws.get(R * D2p_ * 4);
  u16*   mA    = (u16*)ws.get(R * M8_ * 2);
  u16*   mB    = (u16*)ws.get(R * M8_ * 2);
  float* vbuf  = (float*)ws.get((long long)B_ * M8_ * 4);
  u16*   v_bf  = (u16*)ws.get((long long)B_ * M8_ * 2);
  float* fc1   = (float*)ws.get((long long)B_ * UN_ * 4);

  // ---- 1. weight transpose/convert to bf16 [Np, Kp] ------------------------
  auto tc = [&](const float* src, int K, int N, u16* dst, int Kp, int Np) {
    long long tot = (long long)Np * Kp;
    dim3 g((unsigned)((tot + 255) / 256), 1);
    transpose_cvt_k<<<g, 256, 0, stream>>>(src, 0, K, N, dst, 0, Np, Kp);
  };
  for (int i = 0; i < 4; ++i) {
    tc(Wx1[i], E_,  G_, WxT1[i], Ep_, Gp_);
    tc(Wh1[i], H_,  G_, WhT1[i], Hp_, Gp_);
    tc(Wx2[i], M8_, G_, WxT2[i], M8_, Gp_);
    tc(Wh2[i], H_,  G_, WhT2[i], Hp_, Gp_);
  }
  tc(W1, M8_, UN_, W1T, M8_, UN_);

  // ---- 2. embedding gather -------------------------------------------------
  {
    long long tot = R * Ep_;
    gather_emb_k<<<(unsigned)((tot + 255) / 256), 256, 0, stream>>>(p_index, W_emb, embP);
    gather_emb_k<<<(unsigned)((tot + 255) / 256), 256, 0, stream>>>(h_index, W_emb, embH);
  }

  // ---- 3. layer-1 BiLSTMs (p_f, p_b, h_f, h_b) -----------------------------
  const u16*  l1src[4]  = {embP, embP, embH, embH};
  float*      l1dst[4]  = {pseq, pseq, hseq, hseq};
  const int   l1off[4]  = {0, H_, 0, H_};
  const int   l1rev[4]  = {0, 1, 0, 1};
  for (int i = 0; i < 4; ++i) {
    launch_gemm_lds(stream, l1src[i], Ep_, WxT1[i], Ep_, bb1[i], 0,
                    xz, G_, (int)R, G_, /*mode=*/1);
    run_lstm(stream, xz, WhT1[i], l1dst[i], l1off[i], l1rev[i], /*relu=*/l1rev[i],
             z, hping, hpong, cbuf);
  }

  // ---- 4. attention --------------------------------------------------------
  {
    long long tot = R * D2p_;
    cvt_pad_k<<<(unsigned)((tot + 255) / 256), 256, 0, stream>>>(pseq, D2_, p_bf, D2_, D2p_, tot);
    cvt_pad_k<<<(unsigned)((tot + 255) / 256), 256, 0, stream>>>(hseq, D2_, h_bf, D2_, D2p_, tot);
  }
  launch_gemm(stream, p_bf, (long long)T_ * D2p_, D2p_, h_bf, (long long)T_ * D2p_,
              nullptr, 0, 0, emat, (long long)T_ * T_, T_, T_, T_, D2p_, 0, B_);
  softmax_aw_k<<<B_ * T_, T_, 0, stream>>>(emat, h_index, aw);
  softmax_bw_k<<<B_ * T_, T_, 0, stream>>>(emat, p_index, bw);
  {
    long long tot = (long long)B_ * T_ * T_;
    cvt_pad_k<<<(unsigned)((tot + 255) / 256), 256, 0, stream>>>(aw, T_, aw_bf, T_, T_, tot);
    cvt_pad_k<<<(unsigned)((tot + 255) / 256), 256, 0, stream>>>(bw, T_, bw_bf, T_, T_, tot);
  }
  {
    long long tot = (long long)D2p_ * T_;
    dim3 g((unsigned)((tot + 255) / 256), B_);
    transpose_cvt_k<<<g, 256, 0, stream>>>(hseq, (long long)T_ * D2_, T_, D2_,
                                           hT_bf, (long long)D2p_ * T_, D2p_, T_);
    transpose_cvt_k<<<g, 256, 0, stream>>>(pseq, (long long)T_ * D2_, T_, D2_,
                                           pT_bf, (long long)D2p_ * T_, D2p_, T_);
  }
  launch_gemm(stream, aw_bf, (long long)T_ * T_, T_, hT_bf, (long long)D2p_ * T_,
              nullptr, 0, 0, aMat, (long long)T_ * D2p_, D2p_, T_, D2p_, T_, 0, B_);
  launch_gemm(stream, bw_bf, (long long)T_ * T_, T_, pT_bf, (long long)D2p_ * T_,
              nullptr, 0, 0, bMat, (long long)T_ * D2p_, D2p_, T_, D2p_, T_, 0, B_);

  // ---- 5. m_a / m_b --------------------------------------------------------
  {
    long long tot = R * D2_;
    build_m_k<<<(unsigned)((tot + 255) / 256), 256, 0, stream>>>(aMat, pseq, mA);
    build_m_k<<<(unsigned)((tot + 255) / 256), 256, 0, stream>>>(bMat, hseq, mB);
  }

  // ---- 6. layer-2 BiLSTMs (a_f, a_b, b_f, b_b) -----------------------------
  const u16*  l2src[4] = {mA, mA, mB, mB};
  float*      l2dst[4] = {a2seq, a2seq, b2seq, b2seq};
  for (int i = 0; i < 4; ++i) {
    launch_gemm_lds(stream, l2src[i], M8_, WxT2[i], M8_, bb2[i], 0,
                    xz, G_, (int)R, G_, /*mode=*/1);
    run_lstm(stream, xz, WhT2[i], l2dst[i], l1off[i], l1rev[i], /*relu=*/l1rev[i],
             z, hping, hpong, cbuf);
  }

  // ---- 7. pooling + FC head ------------------------------------------------
  pool_meanmax_k<<<(B_ * D2_ + 255) / 256, 256, 0, stream>>>(a2seq, vbuf, 0);
  pool_meanmax_k<<<(B_ * D2_ + 255) / 256, 256, 0, stream>>>(b2seq, vbuf, 2 * D2_);
  {
    long long tot = (long long)B_ * M8_;
    cvt_pad_k<<<(unsigned)((tot + 255) / 256), 256, 0, stream>>>(vbuf, M8_, v_bf, M8_, M8_, tot);
  }
  launch_gemm_lds(stream, v_bf, M8_, W1T, M8_, b1, 0,
                  fc1, UN_, B_, UN_, /*mode=*/3);
  fc2_k<<<1, 256, 0, stream>>>(fc1, W2, b2v, out);
}
